// SOCA_5222680232400
// MI455X (gfx1250) — matrix operand order repaired and checked by reading, not compiled, not run on hardware
//
#include <hip/hip_runtime.h>
#include <hip/hip_bf16.h>
#include <math.h>

typedef __bf16 bf16_t;
typedef __attribute__((ext_vector_type(16))) __bf16 v16bf;
typedef __attribute__((ext_vector_type(8)))  float  v8f;
typedef __attribute__((ext_vector_type(4)))  int    v4i;
typedef __attribute__((address_space(1))) v4i* gptr_v4i;   // global int4*
typedef __attribute__((address_space(3))) v4i* lptr_v4i;   // LDS int4*

#define B_DIM 32
#define C_DIM 256
#define M_DIM 4096

#define MODE_PLAIN 0
#define MODE_T     1

#if defined(__has_builtin)
# if __has_builtin(__builtin_amdgcn_global_load_async_to_lds_b128)
#  define SOCA_HAVE_ASYNC 1
# endif
#endif
#ifndef SOCA_HAVE_ASYNC
# define SOCA_HAVE_ASYNC 0
#endif

// ---------------- per-channel spatial mean ----------------
__global__ void soca_mean_kernel(const float* __restrict__ x, float* __restrict__ meanv) {
    __shared__ float red[256];
    int bc = blockIdx.x;  // b*256 + c
    const float4* row = reinterpret_cast<const float4*>(x + (size_t)bc * M_DIM);
    float s = 0.f;
    for (int i = threadIdx.x; i < M_DIM / 4; i += 256) {
        float4 f = row[i];
        s += f.x + f.y + f.z + f.w;
    }
    red[threadIdx.x] = s;
    __syncthreads();
    for (int st = 128; st > 0; st >>= 1) {
        if (threadIdx.x < st) red[threadIdx.x] += red[threadIdx.x + st];
        __syncthreads();
    }
    if (threadIdx.x == 0) meanv[bc] = red[0] * (1.f / M_DIM);
}

// ---------------- covariance via WMMA bf16 (Gram - mu mu^T) ----------------
// grid: (4 tiles of 128x128, B); block: 256 threads = 8 waves (2x4)
// Async global->LDS copy of f32 tiles (no VGPR staging), f32->bf16 convert
// during fragment build; double-buffered, one barrier per K-chunk.
__global__ void soca_cov_kernel(const float* __restrict__ x,
                                const float* __restrict__ meanv,
                                float* __restrict__ cov) {
    const int b  = blockIdx.y;
    const int tm = blockIdx.x >> 1, tn = blockIdx.x & 1;
    const int rbase = tm * 128, cbase = tn * 128;
    const int tid = threadIdx.x, lane = tid & 31, w = tid >> 5;
    const int wm = w >> 2, wn = w & 3;
    const int waveRow = wm * 64, waveCol = wn * 32;
    const int g = lane >> 4, l15 = lane & 15;
    const float* xb = x + (size_t)b * C_DIM * M_DIM;

    v8f acc[4][2] = {};
    const int NCH = M_DIM / 32;   // 128 chunks

#if SOCA_HAVE_ASYNC
    __shared__ float sAf[2][128][36];   // f32 tiles, row stride 144B (16B aligned)
    __shared__ float sBf[2][128][36];

    auto loadTiles_async = [&](int k0, int bi) {
        #pragma unroll
        for (int p = 0; p < 4; ++p) {
            int c = tid + p * 256;           // 1024 x 16B chunks per tile
            int row = c >> 3, seg = c & 7;
            const float* ga = xb + (size_t)(rbase + row) * M_DIM + k0 + seg * 4;
            const float* gb = xb + (size_t)(cbase + row) * M_DIM + k0 + seg * 4;
            __builtin_amdgcn_global_load_async_to_lds_b128(
                (gptr_v4i)ga, (lptr_v4i)(&sAf[bi][row][seg * 4]), 0, 0);
            __builtin_amdgcn_global_load_async_to_lds_b128(
                (gptr_v4i)gb, (lptr_v4i)(&sBf[bi][row][seg * 4]), 0, 0);
        }
    };

    loadTiles_async(0, 0);
    asm volatile("s_wait_asynccnt 0x0" ::: "memory");
    __syncthreads();

    for (int kc = 0; kc < NCH; ++kc) {
        const int cur = kc & 1, nxt = cur ^ 1;
        const bool hasNext = (kc + 1) < NCH;
        if (hasNext) loadTiles_async((kc + 1) * 32, nxt);

        v16bf afr[4], bfr[2];
        #pragma unroll
        for (int rt = 0; rt < 4; ++rt) {
            int m = waveRow + rt * 16 + l15;
            #pragma unroll
            for (int j = 0; j < 16; ++j) {
                int kk = g * 16 + j;
                afr[rt][j] = (bf16_t)sAf[cur][m][kk];
            }
        }
        #pragma unroll
        for (int ct = 0; ct < 2; ++ct) {
            int n = waveCol + ct * 16 + l15;
            #pragma unroll
            for (int j = 0; j < 16; ++j)
                bfr[ct][j] = (bf16_t)sBf[cur][n][g * 16 + j];   // B[k][n] = x[n][k]
        }
        #pragma unroll
        for (int rt = 0; rt < 4; ++rt)
            #pragma unroll
            for (int ct = 0; ct < 2; ++ct)
                acc[rt][ct] = __builtin_amdgcn_wmma_f32_16x16x32_bf16(
                    false, afr[rt], false, bfr[ct], (short)0, acc[rt][ct], false, false);

        if (hasNext) asm volatile("s_wait_asynccnt 0x0" ::: "memory");
        __syncthreads();
    }
#else
    __shared__ bf16_t sA[2][128][48];
    __shared__ bf16_t sB[2][128][48];

    auto loadRegs = [&](int k0, float4* fa, float4* fb) {
        #pragma unroll
        for (int p = 0; p < 4; ++p) {
            int c = tid + p * 256;
            int row = c >> 3, seg = c & 7;
            fa[p] = reinterpret_cast<const float4*>(xb + (size_t)(rbase + row) * M_DIM + k0)[seg];
            fb[p] = reinterpret_cast<const float4*>(xb + (size_t)(cbase + row) * M_DIM + k0)[seg];
        }
    };
    auto storeTiles = [&](int bi, const float4* fa, const float4* fb) {
        #pragma unroll
        for (int p = 0; p < 4; ++p) {
            int c = tid + p * 256;
            int row = c >> 3, seg = c & 7;
            int kk = seg * 4;
            sA[bi][row][kk + 0] = (bf16_t)fa[p].x; sA[bi][row][kk + 1] = (bf16_t)fa[p].y;
            sA[bi][row][kk + 2] = (bf16_t)fa[p].z; sA[bi][row][kk + 3] = (bf16_t)fa[p].w;
            sB[bi][row][kk + 0] = (bf16_t)fb[p].x; sB[bi][row][kk + 1] = (bf16_t)fb[p].y;
            sB[bi][row][kk + 2] = (bf16_t)fb[p].z; sB[bi][row][kk + 3] = (bf16_t)fb[p].w;
        }
    };

    float4 fa[4], fb[4];
    loadRegs(0, fa, fb);
    storeTiles(0, fa, fb);
    __syncthreads();

    for (int kc = 0; kc < NCH; ++kc) {
        const int cur = kc & 1, nxt = cur ^ 1;
        const bool hasNext = (kc + 1) < NCH;
        if (hasNext) loadRegs((kc + 1) * 32, fa, fb);

        v16bf afr[4], bfr[2];
        #pragma unroll
        for (int rt = 0; rt < 4; ++rt) {
            int m = waveRow + rt * 16 + l15;
            #pragma unroll
            for (int j = 0; j < 16; ++j) {
                int kk = g * 16 + j;
                afr[rt][j] = sA[cur][m][kk];
            }
        }
        #pragma unroll
        for (int ct = 0; ct < 2; ++ct) {
            int n = waveCol + ct * 16 + l15;
            #pragma unroll
            for (int j = 0; j < 16; ++j)
                bfr[ct][j] = sB[cur][n][g * 16 + j];
        }
        #pragma unroll
        for (int rt = 0; rt < 4; ++rt)
            #pragma unroll
            for (int ct = 0; ct < 2; ++ct)
                acc[rt][ct] = __builtin_amdgcn_wmma_f32_16x16x32_bf16(
                    false, afr[rt], false, bfr[ct], (short)0, acc[rt][ct], false, false);

        if (hasNext) storeTiles(nxt, fa, fb);
        __syncthreads();
    }
#endif

    for (int rt = 0; rt < 4; ++rt)
        for (int ct = 0; ct < 2; ++ct)
            for (int vv = 0; vv < 8; ++vv) {
                int grow = rbase + waveRow + rt * 16 + vv + 8 * g;
                int gcol = cbase + waveCol + ct * 16 + l15;
                float val = acc[rt][ct][vv] * (1.f / M_DIM)
                          - meanv[b * C_DIM + grow] * meanv[b * C_DIM + gcol];
                cov[((size_t)b * C_DIM + grow) * C_DIM + gcol] = val;
            }
}

// ---------------- trace ----------------
__global__ void soca_trace_kernel(const float* __restrict__ cov, float* __restrict__ normA) {
    __shared__ float red[256];
    int b = blockIdx.x, t = threadIdx.x;
    red[t] = cov[((size_t)b * C_DIM + t) * C_DIM + t];
    __syncthreads();
    for (int st = 128; st > 0; st >>= 1) {
        if (t < st) red[t] += red[t + st];
        __syncthreads();
    }
    if (t == 0) normA[b] = red[0];
}

// ---------------- Y0 = A/tr(A) (bf16), Z0 = I (bf16) ----------------
__global__ void soca_init_kernel(const float* __restrict__ cov, const float* __restrict__ normA,
                                 bf16_t* __restrict__ Y, bf16_t* __restrict__ Z) {
    size_t idx = (size_t)blockIdx.x * 256 + threadIdx.x;  // 32*65536 elements
    int b  = (int)(idx >> 16);
    int rc = (int)(idx & 65535);
    float inv = 1.f / normA[b];
    Y[idx] = (bf16_t)(cov[idx] * inv);
    Z[idx] = (bf16_t)(((rc >> 8) == (rc & 255)) ? 1.f : 0.f);
}

// ---------------- batched 256x256x256 GEMM (bf16 WMMA) ----------------
// mode PLAIN: D = A*B        mode T: D = 1.5*I - 0.5*(A*B)
// B tile stored transposed in LDS so every fragment read is ds_load_b128.
// Double-buffered; A tile uses async global->LDS copy when available.
__global__ void soca_gemm_kernel(const bf16_t* __restrict__ A,
                                 const bf16_t* __restrict__ Bm,
                                 bf16_t* __restrict__ D, int mode) {
    __shared__ bf16_t sA[2][128][48];
    __shared__ bf16_t sB[2][128][48];   // transposed: [n][k]
    const int b  = blockIdx.y;
    const int tm = blockIdx.x >> 1, tn = blockIdx.x & 1;
    const int rbase = tm * 128, cbase = tn * 128;
    const int tid = threadIdx.x, lane = tid & 31, w = tid >> 5;
    const int wm = w >> 2, wn = w & 3;
    const int waveRow = wm * 64, waveCol = wn * 32;
    const int g = lane >> 4, l15 = lane & 15;
    const bf16_t* Ab = A  + (size_t)b * C_DIM * C_DIM;
    const bf16_t* Bb = Bm + (size_t)b * C_DIM * C_DIM;

#if SOCA_HAVE_ASYNC
    auto loadA_async = [&](int k0, int bi) {
        #pragma unroll
        for (int p = 0; p < 2; ++p) {
            int c = tid + p * 256;           // A: 128x32 bf16 = 512 x 16B chunks
            int row = c >> 2, seg = c & 3;
            const bf16_t* gp = Ab + (size_t)(rbase + row) * C_DIM + k0 + seg * 8;
            __builtin_amdgcn_global_load_async_to_lds_b128(
                (gptr_v4i)gp, (lptr_v4i)(&sA[bi][row][seg * 8]), 0, 0);
        }
    };
#else
    auto loadA_regs = [&](int k0, uint4* ar) {
        #pragma unroll
        for (int p = 0; p < 2; ++p) {
            int c = tid + p * 256;
            int row = c >> 2, seg = c & 3;
            ar[p] = reinterpret_cast<const uint4*>(Ab + (size_t)(rbase + row) * C_DIM + k0)[seg];
        }
    };
    auto storeA = [&](int bi, const uint4* ar) {
        #pragma unroll
        for (int p = 0; p < 2; ++p) {
            int c = tid + p * 256;
            int row = c >> 2, seg = c & 3;
            *reinterpret_cast<uint4*>(&sA[bi][row][seg * 8]) = ar[p];
        }
    };
#endif
    auto loadB_regs = [&](int k0, uint4* br) {
        #pragma unroll
        for (int p = 0; p < 2; ++p) {
            int c = tid + p * 256;           // B: 32x128 bf16 = 512 x 16B chunks
            int row = c >> 4, seg = c & 15;
            br[p] = reinterpret_cast<const uint4*>(Bb + (size_t)(k0 + row) * C_DIM + cbase)[seg];
        }
    };
    auto storeB_tr = [&](int bi, const uint4* br) {
        #pragma unroll
        for (int p = 0; p < 2; ++p) {
            int c = tid + p * 256;
            int row = c >> 4, seg = c & 15;  // row = k, seg*8.. = n
            union { uint4 u; bf16_t h[8]; } cv;
            cv.u = br[p];
            #pragma unroll
            for (int i = 0; i < 8; ++i) sB[bi][seg * 8 + i][row] = cv.h[i];
        }
    };

    v8f acc[4][2] = {};
    uint4 breg[2];
#if !SOCA_HAVE_ASYNC
    uint4 areg[2];
#endif

    // preload chunk 0
#if SOCA_HAVE_ASYNC
    loadA_async(0, 0);
#else
    loadA_regs(0, areg);
#endif
    loadB_regs(0, breg);
#if !SOCA_HAVE_ASYNC
    storeA(0, areg);
#endif
    storeB_tr(0, breg);
#if SOCA_HAVE_ASYNC
    asm volatile("s_wait_asynccnt 0x0" ::: "memory");
#endif
    __syncthreads();

    const int NCH = C_DIM / 32;   // 8 chunks
    for (int kc = 0; kc < NCH; ++kc) {
        const int cur = kc & 1, nxt = cur ^ 1;
        const bool hasNext = (kc + 1) < NCH;
        if (hasNext) {
#if SOCA_HAVE_ASYNC
            loadA_async((kc + 1) * 32, nxt);
#else
            loadA_regs((kc + 1) * 32, areg);
#endif
            loadB_regs((kc + 1) * 32, breg);
        }

        v16bf afr[4], bfr[2];
        #pragma unroll
        for (int rt = 0; rt < 4; ++rt) {
            int m = waveRow + rt * 16 + l15;
            #pragma unroll
            for (int j = 0; j < 16; ++j) {
                int kk = g * 16 + j;
                afr[rt][j] = sA[cur][m][kk];
            }
        }
        #pragma unroll
        for (int ct = 0; ct < 2; ++ct) {
            int n = waveCol + ct * 16 + l15;
            #pragma unroll
            for (int j = 0; j < 16; ++j)
                bfr[ct][j] = sB[cur][n][g * 16 + j];   // contiguous: transposed tile
        }
        #pragma unroll
        for (int rt = 0; rt < 4; ++rt)
            #pragma unroll
            for (int ct = 0; ct < 2; ++ct)
                acc[rt][ct] = __builtin_amdgcn_wmma_f32_16x16x32_bf16(
                    false, afr[rt], false, bfr[ct], (short)0, acc[rt][ct], false, false);

        if (hasNext) {
#if !SOCA_HAVE_ASYNC
            storeA(nxt, areg);
#endif
            storeB_tr(nxt, breg);
#if SOCA_HAVE_ASYNC
            asm volatile("s_wait_asynccnt 0x0" ::: "memory");
#endif
        }
        __syncthreads();
    }

    bf16_t* Db = D + (size_t)b * C_DIM * C_DIM;
    for (int rt = 0; rt < 4; ++rt)
        for (int ct = 0; ct < 2; ++ct)
            for (int vv = 0; vv < 8; ++vv) {
                int grow = rbase + waveRow + rt * 16 + vv + 8 * g;
                int gcol = cbase + waveCol + ct * 16 + l15;
                float val = acc[rt][ct][vv];
                if (mode == MODE_T) val = ((grow == gcol) ? 1.5f : 0.f) - 0.5f * val;
                Db[(size_t)grow * C_DIM + gcol] = (bf16_t)val;
            }
}

// ---------------- v = sqrt(normA) * column-mean(Y) ----------------
__global__ void soca_rowmean_kernel(const bf16_t* __restrict__ Y,
                                    const float* __restrict__ normA,
                                    float* __restrict__ vvec) {
    int b = blockIdx.x, c = threadIdx.x;
    const bf16_t* Yb = Y + (size_t)b * C_DIM * C_DIM;
    float s = 0.f;
    for (int r = 0; r < C_DIM; ++r) s += (float)Yb[(size_t)r * C_DIM + c];
    vvec[b * C_DIM + c] = s * (1.f / C_DIM) * sqrtf(normA[b]);
}

// ---------------- FC -> relu -> FC -> sigmoid ----------------
__global__ void soca_fc_kernel(const float* __restrict__ vvec,
                               const float* __restrict__ w1, const float* __restrict__ b1,
                               const float* __restrict__ w2, const float* __restrict__ b2,
                               float* __restrict__ gate) {
    __shared__ float vs[256];
    __shared__ float hs[32];
    int b = blockIdx.x, t = threadIdx.x;
    vs[t] = vvec[b * 256 + t];
    __syncthreads();
    if (t < 32) {
        float s = b1[t];
        for (int c = 0; c < 256; ++c) s += w1[t * 256 + c] * vs[c];
        hs[t] = fmaxf(s, 0.f);
    }
    __syncthreads();
    float s = b2[t];
    for (int o = 0; o < 32; ++o) s += w2[t * 32 + o] * hs[o];
    gate[b * 256 + t] = 1.f / (1.f + __expf(-s));
}

// ---------------- out = gate[b,c] * x ----------------
__global__ void soca_scale_kernel(const float* __restrict__ x,
                                  const float* __restrict__ gate,
                                  float* __restrict__ out) {
    size_t idx = (size_t)blockIdx.x * 256 + threadIdx.x;  // float4 index
    int chan = (int)(idx >> 10);                          // 1024 float4 per channel
    float gval = gate[chan];
    float4 f = reinterpret_cast<const float4*>(x)[idx];
    f.x *= gval; f.y *= gval; f.z *= gval; f.w *= gval;
    reinterpret_cast<float4*>(out)[idx] = f;
}

extern "C" void kernel_launch(void* const* d_in, const int* in_sizes, int n_in,
                              void* d_out, int out_size, void* d_ws, size_t ws_size,
                              hipStream_t stream) {
    (void)in_sizes; (void)n_in; (void)out_size; (void)ws_size;
    const float* x  = (const float*)d_in[0];
    const float* w1 = (const float*)d_in[1];
    const float* b1 = (const float*)d_in[2];
    const float* w2 = (const float*)d_in[3];
    const float* b2 = (const float*)d_in[4];
    float* out = (float*)d_out;

    char* ws = (char*)d_ws;
    size_t off = 0;
    auto walloc = [&](size_t bytes) -> void* {
        void* p = ws + off;
        off += (bytes + 255) & ~(size_t)255;
        return p;
    };

    const size_t mat_f32 = (size_t)B_DIM * C_DIM * C_DIM * sizeof(float);
    const size_t mat_bf  = (size_t)B_DIM * C_DIM * C_DIM * sizeof(bf16_t);
    float*  cov   = (float*) walloc(mat_f32);
    bf16_t* Ybuf0 = (bf16_t*)walloc(mat_bf);
    bf16_t* Ybuf1 = (bf16_t*)walloc(mat_bf);
    bf16_t* Zbuf0 = (bf16_t*)walloc(mat_bf);
    bf16_t* Zbuf1 = (bf16_t*)walloc(mat_bf);
    bf16_t* Tbuf  = (bf16_t*)walloc(mat_bf);
    float* meanv = (float*)walloc(B_DIM * C_DIM * sizeof(float));
    float* normA = (float*)walloc(B_DIM * sizeof(float));
    float* vvec  = (float*)walloc(B_DIM * C_DIM * sizeof(float));
    float* gate  = (float*)walloc(B_DIM * C_DIM * sizeof(float));

    soca_mean_kernel <<<dim3(B_DIM * C_DIM), 256, 0, stream>>>(x, meanv);
    soca_cov_kernel  <<<dim3(4, B_DIM),      256, 0, stream>>>(x, meanv, cov);
    soca_trace_kernel<<<dim3(B_DIM),         256, 0, stream>>>(cov, normA);
    soca_init_kernel <<<dim3(8192),          256, 0, stream>>>(cov, normA, Ybuf0, Zbuf0);

    bf16_t* Y[2] = {Ybuf0, Ybuf1};
    bf16_t* Z[2] = {Zbuf0, Zbuf1};
    int cur = 0;
    for (int i = 0; i < 5; ++i) {
        int nxt = cur ^ 1;
        soca_gemm_kernel<<<dim3(4, B_DIM), 256, 0, stream>>>(Z[cur], Y[cur], Tbuf, MODE_T);
        soca_gemm_kernel<<<dim3(4, B_DIM), 256, 0, stream>>>(Y[cur], Tbuf, Y[nxt], MODE_PLAIN);
        if (i < 4)
            soca_gemm_kernel<<<dim3(4, B_DIM), 256, 0, stream>>>(Tbuf, Z[cur], Z[nxt], MODE_PLAIN);
        cur = nxt;
    }

    soca_rowmean_kernel<<<dim3(B_DIM), 256, 0, stream>>>(Y[cur], normA, vvec);
    soca_fc_kernel     <<<dim3(B_DIM), 256, 0, stream>>>(vvec, w1, b1, w2, b2, gate);
    soca_scale_kernel  <<<dim3(32768), 256, 0, stream>>>(x, gate, out);
}
